// KimiDeltaAttention_6339371729798
// MI455X (gfx1250) — matrix operand order repaired and checked by reading, not compiled, not run on hardware
//
#include <hip/hip_runtime.h>
#include <hip/hip_bf16.h>

// ---- problem dims (compile-time) ----
constexpr int T_   = 4096;
constexpr int HID_ = 2048;
constexpr int H_   = 16;
constexpr int D_   = 128;
constexpr int KC_  = 4;       // conv kernel
constexpr int HD_  = H_ * D_; // 2048
constexpr float EPS_ = 1e-6f;

// ---- WMMA vector types (probe-verified signatures) ----
typedef __attribute__((ext_vector_type(16))) __bf16 v16bf;
typedef __attribute__((ext_vector_type(8)))  __bf16 v8bf;
typedef __attribute__((ext_vector_type(8)))  float  v8f;

__device__ __forceinline__ __bf16 f2bf(float f) {
  union { float f; unsigned u; } a; a.f = f;
  unsigned r = a.u + 0x7FFFu + ((a.u >> 16) & 1u);   // RNE
  unsigned short h = (unsigned short)(r >> 16);
  __bf16 b;
  __builtin_memcpy(&b, &h, 2);
  return b;
}
__device__ __forceinline__ float sigmoidf_(float x) { return 1.f / (1.f + __expf(-x)); }
__device__ __forceinline__ float siluf_(float x)    { return x * sigmoidf_(x); }

// ---------------------------------------------------------------------------
// fp32 -> bf16 convert
// ---------------------------------------------------------------------------
__global__ void cvt_f32_bf16_kernel(const float* __restrict__ in,
                                    __bf16* __restrict__ out, int n) {
  int i = blockIdx.x * 256 + threadIdx.x;
  if (i < n) out[i] = f2bf(in[i]);
}

// ---------------------------------------------------------------------------
// bf16 WMMA GEMM: C[M,N] = A[M,K] @ B[K,N], fp32 accumulate.
// 256 threads (8 waves), tile 128x128, K-step 32, double-buffered LDS with
// register-staged global prefetch: one barrier per K-iteration; next tile's
// global loads are issued before the current tile's 8 WMMAs so VMEM overlaps
// the matrix pipe. global_prefetch_b8 warms L2 two tiles ahead.
// ---------------------------------------------------------------------------
__global__ __launch_bounds__(256)
void wmma_gemm_bf16_kernel(const __bf16* __restrict__ A,
                           const __bf16* __restrict__ B,
                           float* __restrict__ C, int M, int N, int Kd) {
  constexpr int TM = 128, TN = 128, TK = 32, LP = 40; // 80B LDS rows (16B aligned)
  __shared__ __align__(16) __bf16 As[2][TM][LP]; // [buf][m][k]
  __shared__ __align__(16) __bf16 Bs[2][TN][LP]; // [buf][n][k] (transposed)

  const int bm = blockIdx.x * TM;
  const int bn = blockIdx.y * TN;
  const int tid  = threadIdx.x;
  const int wave = tid >> 5;
  const int lane = tid & 31;
  const int lhalf = lane >> 4;   // 0 | 1
  const int lmod  = lane & 15;
  const int wm = (wave >> 2) * 64;
  const int wn = (wave & 3)  * 32;

  // per-thread staging coordinates (2 chunks of A, 2 chunks of B per thread)
  const int am0 = tid >> 2,       ak0 = (tid & 3) << 3;
  const int am1 = am0 + 64,       ak1 = ak0;
  const int bk0 = tid >> 4,       bnn = (tid & 15) << 3;
  const int bk1 = bk0 + 16;

  uint4 ar0, ar1, br0, br1;
  auto loadG = [&](int k0) {
    ar0 = *(const uint4*)&A[(size_t)(bm + am0) * Kd + k0 + ak0];
    ar1 = *(const uint4*)&A[(size_t)(bm + am1) * Kd + k0 + ak1];
    br0 = *(const uint4*)&B[(size_t)(k0 + bk0) * N + bn + bnn];
    br1 = *(const uint4*)&B[(size_t)(k0 + bk1) * N + bn + bnn];
  };
  auto storeL = [&](int buf) {
    *(uint4*)&As[buf][am0][ak0] = ar0;
    *(uint4*)&As[buf][am1][ak1] = ar1;
    union { uint4 u; __bf16 b[8]; } t0, t1;
    t0.u = br0; t1.u = br1;
    #pragma unroll
    for (int j = 0; j < 8; ++j) Bs[buf][bnn + j][bk0] = t0.b[j];
    #pragma unroll
    for (int j = 0; j < 8; ++j) Bs[buf][bnn + j][bk1] = t1.b[j];
  };

  v8f zero;
  #pragma unroll
  for (int i = 0; i < 8; ++i) zero[i] = 0.f;
  v8f acc[4][2];
  #pragma unroll
  for (int r = 0; r < 4; ++r)
    #pragma unroll
    for (int c = 0; c < 2; ++c) acc[r][c] = zero;

  const int nIter = Kd / TK;
  loadG(0);
  for (int it = 0; it < nIter; ++it) {
    const int buf = it & 1;
    storeL(buf);
    __syncthreads();
    if (it + 1 < nIter) loadG((it + 1) * TK);   // overlap with WMMAs below
    if (it + 2 < nIter) {
      const int k2 = (it + 2) * TK;
      __builtin_prefetch(&A[(size_t)(bm + am0) * Kd + k2 + ak0], 0, 1);
      __builtin_prefetch(&B[(size_t)(k2 + bk0) * N + bn + bnn], 0, 1);
    }

    // Fragments per ISA 16-bit layouts:
    // A: lane<16 -> K {0..7,16..23}; lane>=16 -> K {8..15,24..31}; M = lane%16
    // B: lane<16 -> K {0..7,16..23}; lane>=16 -> K {8..15,24..31}; N = lane%16
    v16bf af[4], bfr[2];
    #pragma unroll
    for (int r = 0; r < 4; ++r) {
      const __bf16* ap = &As[buf][wm + r * 16 + lmod][lhalf * 8];
      union { v16bf v; v8bf h[2]; } u;
      u.h[0] = *(const v8bf*)(ap);
      u.h[1] = *(const v8bf*)(ap + 16);
      af[r] = u.v;
    }
    #pragma unroll
    for (int c = 0; c < 2; ++c) {
      const __bf16* bp = &Bs[buf][wn + c * 16 + lmod][lhalf * 8];
      union { v16bf v; v8bf h[2]; } u;
      u.h[0] = *(const v8bf*)(bp);
      u.h[1] = *(const v8bf*)(bp + 16);
      bfr[c] = u.v;
    }
    #pragma unroll
    for (int r = 0; r < 4; ++r)
      #pragma unroll
      for (int c = 0; c < 2; ++c)
        acc[r][c] = __builtin_amdgcn_wmma_f32_16x16x32_bf16(
            false, af[r], false, bfr[c], (short)0, acc[r][c], false, false);
    // no trailing barrier: buffer `buf` is only overwritten at it+2, after the
    // barrier at it+1 has proven all waves finished reading it.
  }

  // D layout: VGPR rr, lane: M = rr + 8*(lane/16), N = lane%16
  #pragma unroll
  for (int r = 0; r < 4; ++r)
    #pragma unroll
    for (int c = 0; c < 2; ++c)
      #pragma unroll
      for (int rr = 0; rr < 8; ++rr) {
        int row = bm + wm + r * 16 + lhalf * 8 + rr;
        int col = bn + wn + c * 16 + lmod;
        C[(size_t)row * N + col] = acc[r][c][rr];
      }
}

// ---------------------------------------------------------------------------
// beta[t,h] = sigmoid(x[t,:] @ Wb[:,h]);  grid=T, block=128 (8 lanes per head)
// ---------------------------------------------------------------------------
__global__ __launch_bounds__(128)
void beta_kernel(const float* __restrict__ x, const float* __restrict__ Wb,
                 float* __restrict__ beta) {
  int t = blockIdx.x, tid = threadIdx.x;
  int h = tid >> 3, s = tid & 7;
  float acc = 0.f;
  for (int k = s; k < HID_; k += 8)
    acc += x[(size_t)t * HID_ + k] * Wb[(size_t)k * H_ + h];
  __shared__ float r[128];
  r[tid] = acc; __syncthreads();
  if (s < 4) r[tid] += r[tid + 4]; __syncthreads();
  if (s < 2) r[tid] += r[tid + 2]; __syncthreads();
  if (s < 1) beta[t * H_ + h] = sigmoidf_(r[tid] + r[tid + 1]);
}

// ---------------------------------------------------------------------------
// causal conv(K=4) + SiLU for q/k/v; l2norm(q,k) per (t,h); decay g.
// grid=(T,H), block=128 (one thread per head-dim channel).
// ---------------------------------------------------------------------------
__global__ __launch_bounds__(128)
void conv_act_kernel(const float* __restrict__ qpre, const float* __restrict__ kpre,
                     const float* __restrict__ vpre,
                     const float* __restrict__ wqc, const float* __restrict__ wkc,
                     const float* __restrict__ wvc,
                     const float* __restrict__ f2, const float* __restrict__ dt_bias,
                     const float* __restrict__ A_log,
                     float* __restrict__ qo, float* __restrict__ ko,
                     float* __restrict__ vo, float* __restrict__ go) {
  int t = blockIdx.x, h = blockIdx.y, d = threadIdx.x;
  int c = h * D_ + d;
  float qv = 0.f, kv = 0.f, vv = 0.f;
  #pragma unroll
  for (int i = 0; i < KC_; ++i) {
    int tt = t - (KC_ - 1) + i;
    if (tt >= 0) {
      size_t idx = (size_t)tt * HD_ + c;
      qv += wqc[c * KC_ + i] * qpre[idx];
      kv += wkc[c * KC_ + i] * kpre[idx];
      vv += wvc[c * KC_ + i] * vpre[idx];
    }
  }
  qv = siluf_(qv); kv = siluf_(kv); vv = siluf_(vv);

  __shared__ float rq[128], rk[128];
  rq[d] = qv * qv; rk[d] = kv * kv;
  __syncthreads();
  #pragma unroll
  for (int off = 64; off > 0; off >>= 1) {
    if (d < off) { rq[d] += rq[d + off]; rk[d] += rk[d + off]; }
    __syncthreads();
  }
  float qn = qv * rsqrtf(rq[0] + EPS_);
  float kn = kv * rsqrtf(rk[0] + EPS_);

  size_t idx = (size_t)t * HD_ + c;
  float fv = f2[idx] + dt_bias[c];
  float sp = (fv > 20.f) ? fv : log1pf(__expf(fv));   // softplus
  float gv = -__expf(A_log[h]) * sp;

  qo[idx] = qn; ko[idx] = kn; vo[idx] = vv; go[idx] = gv;
}

// ---------------------------------------------------------------------------
// Sequential delta-rule scan. grid=H, block=128 (thread j owns value column j;
// state column S[0..127] lives in VGPRs, fully unrolled). float4 LDS reads and
// 4-way split accumulators keep the VALU near 1 IPC despite 1 wave/SIMD.
// ---------------------------------------------------------------------------
__global__ __launch_bounds__(128)
void scan_kernel(const float* __restrict__ q, const float* __restrict__ k,
                 const float* __restrict__ v, const float* __restrict__ g,
                 const float* __restrict__ beta, float* __restrict__ o) {
  const int h = blockIdx.x, j = threadIdx.x;
  const float scale = 0.08838834764831845f; // D^-0.5
  float S[D_];
  #pragma unroll
  for (int kk = 0; kk < D_; ++kk) S[kk] = 0.f;
  __shared__ __align__(16) float ks[D_], qs[D_], es[D_];

  #pragma unroll 1
  for (int t = 0; t < T_; ++t) {
    size_t base = (size_t)t * HD_ + h * D_;
    __syncthreads();                    // protect prev-iter LDS reads
    ks[j] = k[base + j];
    qs[j] = q[base + j];
    es[j] = __expf(g[base + j]);
    float vj = v[base + j];
    float bt = beta[t * H_ + h];
    __syncthreads();

    float a0 = 0.f, a1 = 0.f, a2 = 0.f, a3 = 0.f;
    #pragma unroll
    for (int kk = 0; kk < D_; kk += 4) {
      float4 e4 = *(const float4*)&es[kk];
      float4 k4 = *(const float4*)&ks[kk];
      S[kk + 0] *= e4.x; S[kk + 1] *= e4.y; S[kk + 2] *= e4.z; S[kk + 3] *= e4.w;
      a0 += S[kk + 0] * k4.x; a1 += S[kk + 1] * k4.y;
      a2 += S[kk + 2] * k4.z; a3 += S[kk + 3] * k4.w;
    }
    float dv = bt * (vj - ((a0 + a1) + (a2 + a3)));

    float o0 = 0.f, o1 = 0.f, o2 = 0.f, o3 = 0.f;
    #pragma unroll
    for (int kk = 0; kk < D_; kk += 4) {
      float4 k4 = *(const float4*)&ks[kk];
      float4 q4 = *(const float4*)&qs[kk];
      S[kk + 0] += k4.x * dv; o0 += S[kk + 0] * q4.x;
      S[kk + 1] += k4.y * dv; o1 += S[kk + 1] * q4.y;
      S[kk + 2] += k4.z * dv; o2 += S[kk + 2] * q4.z;
      S[kk + 3] += k4.w * dv; o3 += S[kk + 3] * q4.w;
    }
    o[base + j] = ((o0 + o1) + (o2 + o3)) * scale;
  }
}

// ---------------------------------------------------------------------------
// gated RMSNorm + SiLU(gate) -> bf16 for the final WMMA GEMM. grid=(T,H).
// ---------------------------------------------------------------------------
__global__ __launch_bounds__(128)
void gate_norm_kernel(const float* __restrict__ o, const float* __restrict__ g2,
                      const float* __restrict__ norm_w, __bf16* __restrict__ onb) {
  int t = blockIdx.x, h = blockIdx.y, d = threadIdx.x;
  size_t idx = (size_t)t * HD_ + h * D_ + d;
  float ov = o[idx];
  __shared__ float r[128];
  r[d] = ov * ov; __syncthreads();
  #pragma unroll
  for (int off = 64; off > 0; off >>= 1) {
    if (d < off) r[d] += r[d + off];
    __syncthreads();
  }
  float inv = rsqrtf(r[0] * (1.f / D_) + EPS_);
  float gt = g2[idx];
  onb[idx] = f2bf(ov * inv * norm_w[d] * siluf_(gt));
}

// ---------------------------------------------------------------------------
// host launcher
// ---------------------------------------------------------------------------
extern "C" void kernel_launch(void* const* d_in, const int* in_sizes, int n_in,
                              void* d_out, int out_size, void* d_ws, size_t ws_size,
                              hipStream_t stream) {
  const float* x       = (const float*)d_in[0];
  const float* Wq      = (const float*)d_in[1];
  const float* Wk      = (const float*)d_in[2];
  const float* Wv      = (const float*)d_in[3];
  const float* wq_conv = (const float*)d_in[4];
  const float* wk_conv = (const float*)d_in[5];
  const float* wv_conv = (const float*)d_in[6];
  const float* A_log   = (const float*)d_in[7];
  const float* dt_bias = (const float*)d_in[8];
  const float* Wfa     = (const float*)d_in[9];
  const float* Wfb     = (const float*)d_in[10];
  const float* Wb      = (const float*)d_in[11];
  const float* Wga     = (const float*)d_in[12];
  const float* Wgb     = (const float*)d_in[13];
  const float* norm_w  = (const float*)d_in[14];
  const float* Wo      = (const float*)d_in[15];
  float* out = (float*)d_out;

  char* p = (char*)d_ws;
  auto bump = [&](size_t bytes) -> void* {
    void* r = (void*)p;
    p += (bytes + 255) & ~(size_t)255;
    return r;
  };

  __bf16* xb   = (__bf16*)bump((size_t)T_ * HID_ * 2);
  __bf16* wqb  = (__bf16*)bump((size_t)HID_ * HD_ * 2);
  __bf16* wkb  = (__bf16*)bump((size_t)HID_ * HD_ * 2);
  __bf16* wvb  = (__bf16*)bump((size_t)HID_ * HD_ * 2);
  __bf16* wob  = (__bf16*)bump((size_t)HD_ * HID_ * 2);
  __bf16* wfab = (__bf16*)bump((size_t)HID_ * D_ * 2);
  __bf16* wfbb = (__bf16*)bump((size_t)D_ * HD_ * 2);
  __bf16* wgab = (__bf16*)bump((size_t)HID_ * D_ * 2);
  __bf16* wgbb = (__bf16*)bump((size_t)D_ * HD_ * 2);
  float*  qpre = (float*)bump((size_t)T_ * HD_ * 4);
  float*  kpre = (float*)bump((size_t)T_ * HD_ * 4);
  float*  vpre = (float*)bump((size_t)T_ * HD_ * 4);
  float*  f1   = (float*)bump((size_t)T_ * D_ * 4);
  __bf16* f1b  = (__bf16*)bump((size_t)T_ * D_ * 2);
  float*  f2   = (float*)bump((size_t)T_ * HD_ * 4);
  float*  g1   = (float*)bump((size_t)T_ * D_ * 4);
  __bf16* g1b  = (__bf16*)bump((size_t)T_ * D_ * 2);
  float*  g2   = (float*)bump((size_t)T_ * HD_ * 4);
  float*  qf   = (float*)bump((size_t)T_ * HD_ * 4);
  float*  kf   = (float*)bump((size_t)T_ * HD_ * 4);
  float*  vf   = (float*)bump((size_t)T_ * HD_ * 4);
  float*  gf   = (float*)bump((size_t)T_ * HD_ * 4);
  float*  beta = (float*)bump((size_t)T_ * H_ * 4);
  float*  of   = (float*)bump((size_t)T_ * HD_ * 4);
  __bf16* onb  = (__bf16*)bump((size_t)T_ * HD_ * 2);

  auto cvt = [&](const float* in, __bf16* outp, size_t n) {
    cvt_f32_bf16_kernel<<<dim3((unsigned)((n + 255) / 256)), 256, 0, stream>>>(
        in, outp, (int)n);
  };
  auto gemm = [&](const __bf16* A, const __bf16* B, float* C, int M, int N, int Kd) {
    wmma_gemm_bf16_kernel<<<dim3(M / 128, N / 128), 256, 0, stream>>>(A, B, C, M, N, Kd);
  };

  // bf16 conversions
  cvt(x,   xb,   (size_t)T_ * HID_);
  cvt(Wq,  wqb,  (size_t)HID_ * HD_);
  cvt(Wk,  wkb,  (size_t)HID_ * HD_);
  cvt(Wv,  wvb,  (size_t)HID_ * HD_);
  cvt(Wo,  wob,  (size_t)HD_ * HID_);
  cvt(Wfa, wfab, (size_t)HID_ * D_);
  cvt(Wfb, wfbb, (size_t)D_ * HD_);
  cvt(Wga, wgab, (size_t)HID_ * D_);
  cvt(Wgb, wgbb, (size_t)D_ * HD_);

  // projections (WMMA)
  gemm(xb, wqb, qpre, T_, HD_, HID_);
  gemm(xb, wkb, kpre, T_, HD_, HID_);
  gemm(xb, wvb, vpre, T_, HD_, HID_);

  // low-rank decay path: f2 = (x@Wfa)@Wfb
  gemm(xb, wfab, f1, T_, D_, HID_);
  cvt(f1, f1b, (size_t)T_ * D_);
  gemm(f1b, wfbb, f2, T_, HD_, D_);

  // low-rank gate path: g2 = (x@Wga)@Wgb
  gemm(xb, wgab, g1, T_, D_, HID_);
  cvt(g1, g1b, (size_t)T_ * D_);
  gemm(g1b, wgbb, g2, T_, HD_, D_);

  // beta
  beta_kernel<<<dim3(T_), 128, 0, stream>>>(x, Wb, beta);

  // conv + SiLU + l2norm + decay
  conv_act_kernel<<<dim3(T_, H_), 128, 0, stream>>>(
      qpre, kpre, vpre, wq_conv, wk_conv, wv_conv, f2, dt_bias, A_log,
      qf, kf, vf, gf);

  // sequential delta-rule scan
  scan_kernel<<<dim3(H_), 128, 0, stream>>>(qf, kf, vf, gf, beta, of);

  // gated RMSNorm -> bf16
  gate_norm_kernel<<<dim3(T_, H_), 128, 0, stream>>>(of, g2, norm_w, onb);

  // output projection (WMMA)
  gemm(onb, wob, out, T_, HID_, HD_);
}
